// ConditionalDecoder_68753836474984
// MI455X (gfx1250) — matrix operand order repaired and checked by reading, not compiled, not run on hardware
//
#include <hip/hip_runtime.h>
#include <hip/hip_bf16.h>

// Problem constants (fixed by the reference).
#define TT 100
#define BB 64
#define SS 128
#define EE 512
#define HH 1024
#define CC 512
#define VV 32000

typedef __attribute__((ext_vector_type(16))) __bf16 bf16x16;
typedef __attribute__((ext_vector_type(8)))  float  floatx8;

union AFrag { bf16x16 v; uint4 q[2]; };
union CFrag { floatx8 v; float f[8]; };

// ---------------------------------------------------------------------------
// Generic bf16 WMMA GEMM:  Cout[M,N] = A[M,K] * W[N,K]^T (+ bias[N])
// A row-major (lda=K), W row-major with output-features as rows (lda=K).
// One wave computes a 16 x (16*NT) tile; the A fragment is loaded once per
// K-chunk and reused by NT WMMAs. K stepped by 32 per wmma.
// VGPR fragment packing follows CDNA5 ISA 7.12.2 layouts (wave32).
//   NT=1: max wave-parallelism for the small-M serial-path GEMMs.
//   NT=4: higher arithmetic intensity for the big vocab / ctx GEMMs.
// ---------------------------------------------------------------------------
template <int NT>
__global__ __launch_bounds__(32) void gemm_bf16_wmma(
    const __bf16* __restrict__ A, const __bf16* __restrict__ W,
    const float* __restrict__ bias, float* __restrict__ Cout,
    int M, int N, int K)
{
  const int lane = threadIdx.x;     // 0..31
  const int half = lane >> 4;       // 0 or 1
  const int l16  = lane & 15;
  const int m0 = blockIdx.y << 4;
  const int n0 = blockIdx.x * (16 * NT);

  const __bf16* Arow  = A + (size_t)(m0 + l16) * K;  // A row m0+l16
  const __bf16* Wrow0 = W + (size_t)(n0 + l16) * K;  // W row (output feature) n0+l16

  CFrag acc[NT];
  #pragma unroll
  for (int j = 0; j < NT; ++j)
    #pragma unroll
    for (int r = 0; r < 8; ++r) acc[j].f[r] = 0.0f;

  for (int k = 0; k < K; k += 32) {
    // A 16x32 fragment: e=0..7 -> K=k+half*8+e ; e=8..15 -> K=k+16+half*8+(e-8)
    AFrag a;
    a.q[0] = *(const uint4*)(Arow + k + half * 8);
    a.q[1] = *(const uint4*)(Arow + k + 16 + half * 8);
    #pragma unroll
    for (int j = 0; j < NT; ++j) {
      const __bf16* Wrow = Wrow0 + (size_t)j * 16 * K;
      // B 32x16 fragment (K-major per column): e=0..7 -> K=k+half*8+e ; e=8..15 -> K=k+16+half*8+(e-8), col=l16
      AFrag b;
      b.q[0] = *(const uint4*)(Wrow + k + half * 8);
      b.q[1] = *(const uint4*)(Wrow + k + 16 + half * 8);
      // Stream next weight chunk toward L0/WGP$ (weights are L2-resident).
      __builtin_prefetch((const void*)(Wrow + k + 256), 0, 1);
      acc[j].v = __builtin_amdgcn_wmma_f32_16x16x32_bf16(
          false, a.v, false, b.v, (short)0, acc[j].v, false, false);
    }
  }

  // D layout: lane -> column (n + l16), VGPR r -> row m0 + half*8 + r
  #pragma unroll
  for (int j = 0; j < NT; ++j) {
    const int n = n0 + j * 16 + l16;
    const float bv = bias ? bias[n] : 0.0f;
    float* crow = Cout + (size_t)(m0 + half * 8) * N + n;
    #pragma unroll
    for (int r = 0; r < 8; ++r)
      crow[(size_t)r * N] = acc[j].f[r] + bv;
  }
}

// ---------------------------------------------------------------------------
// Support kernels
// ---------------------------------------------------------------------------
__global__ void cvt_f32_bf16(const float* __restrict__ s, __bf16* __restrict__ d, int n)
{
  int i = blockIdx.x * blockDim.x + threadIdx.x;
  if (i < n) d[i] = (__bf16)s[i];
}

// Gather y_emb for all decoder steps (rows 0..T-2) straight to bf16.
__global__ void embed_gather(const float* __restrict__ emb, const int* __restrict__ y,
                             __bf16* __restrict__ xall)
{
  const size_t n = (size_t)(TT - 1) * BB * EE;
  size_t i = (size_t)blockIdx.x * blockDim.x + threadIdx.x;
  if (i >= n) return;
  int e = (int)(i % EE);
  int r = (int)(i / EE);               // r = t*BB + b
  int tok = y[r];
  xall[i] = (__bf16)emb[(size_t)tok * EE + e];
}

__global__ void init_state(float* __restrict__ hf, __bf16* __restrict__ hb,
                           float* __restrict__ lossacc)
{
  int i = blockIdx.x * blockDim.x + threadIdx.x;
  if (i < BB * HH) { hf[i] = 0.0f; hb[i] = (__bf16)0.0f; }
  if (i == 0) *lossacc = 0.0f;
}

// Fused GRU gates: consumes gi = x@Wih^T+bih and gh = h@Whh^T+bhh (fp32),
// produces new hidden in fp32 and bf16 (bf16 feeds the next WMMA GEMM).
__global__ void gru_fuse(const float* __restrict__ gi, const float* __restrict__ gh,
                         const float* __restrict__ hprev,
                         float* __restrict__ hout, __bf16* __restrict__ houtb)
{
  int i = blockIdx.x * blockDim.x + threadIdx.x;
  if (i >= BB * HH) return;
  int b = i / HH, j = i % HH;
  const float* gib = gi + (size_t)b * 3 * HH;
  const float* ghb = gh + (size_t)b * 3 * HH;
  float r = 1.0f / (1.0f + __expf(-(gib[j] + ghb[j])));
  float z = 1.0f / (1.0f + __expf(-(gib[HH + j] + ghb[HH + j])));
  float n = tanhf(gib[2 * HH + j] + r * ghb[2 * HH + j]);
  float h = (1.0f - z) * n + z * hprev[i];
  hout[i] = h;
  houtb[i] = (__bf16)h;
}

// scores[s*B+b] = sum_c tanh(ctx_p[s,b,c] + hid[b,c]) * w_mlp[c]
__global__ __launch_bounds__(128) void att_scores(
    const float* __restrict__ ctxp, const float* __restrict__ hid,
    const float* __restrict__ wmlp, float* __restrict__ sc)
{
  int sb = blockIdx.x;                  // s*BB + b
  int b = sb % BB;
  const float* cp = ctxp + (size_t)sb * CC;
  const float* hb = hid + (size_t)b * CC;
  float p = 0.0f;
  for (int c = threadIdx.x; c < CC; c += 128)
    p += tanhf(cp[c] + hb[c]) * wmlp[c];
  __shared__ float red[128];
  red[threadIdx.x] = p; __syncthreads();
  for (int s = 64; s > 0; s >>= 1) {
    if (threadIdx.x < s) red[threadIdx.x] += red[threadIdx.x + s];
    __syncthreads();
  }
  if (threadIdx.x == 0) sc[sb] = red[0];
}

// Per batch column: softmax over S, then z[b,:] = sum_s alpha[s]*ctx[s,b,:] (to bf16).
__global__ __launch_bounds__(128) void att_softmax_z(
    const float* __restrict__ sc, const float* __restrict__ ctx,
    __bf16* __restrict__ zb)
{
  int b = blockIdx.x;
  int tid = threadIdx.x;                // SS == 128 == blockDim.x
  __shared__ float alpha[SS];
  __shared__ float red[SS];
  float v = sc[(size_t)tid * BB + b];
  red[tid] = v; __syncthreads();
  for (int s = 64; s > 0; s >>= 1) {
    if (tid < s) red[tid] = fmaxf(red[tid], red[tid + s]);
    __syncthreads();
  }
  float mx = red[0]; __syncthreads();
  float e = __expf(v - mx);
  red[tid] = e; __syncthreads();
  for (int s = 64; s > 0; s >>= 1) {
    if (tid < s) red[tid] += red[tid + s];
    __syncthreads();
  }
  alpha[tid] = e / red[0]; __syncthreads();
  for (int c = tid; c < CC; c += 128) {
    float acc = 0.0f;
    for (int s = 0; s < SS; ++s)
      acc += alpha[s] * ctx[((size_t)s * BB + b) * CC + c];
    zb[(size_t)b * CC + c] = (__bf16)acc;
  }
}

__global__ void tanh_cast(const float* __restrict__ in, __bf16* __restrict__ out, int n)
{
  int i = blockIdx.x * blockDim.x + threadIdx.x;
  if (i < n) out[i] = (__bf16)tanhf(in[i]);
}

// Row-wise log-softmax over V + target NLL, accumulated into *acc.
__global__ __launch_bounds__(256) void loss_kernel(
    const float* __restrict__ sbuf, const int* __restrict__ y,
    int tnext, float* __restrict__ acc)
{
  int b = blockIdx.x;
  const float* row = sbuf + (size_t)b * VV;
  __shared__ float red[256];
  float m = -3.4e38f;
  for (int v = threadIdx.x; v < VV; v += 256) m = fmaxf(m, row[v]);
  red[threadIdx.x] = m; __syncthreads();
  for (int s = 128; s > 0; s >>= 1) {
    if (threadIdx.x < s) red[threadIdx.x] = fmaxf(red[threadIdx.x], red[threadIdx.x + s]);
    __syncthreads();
  }
  float mx = red[0]; __syncthreads();
  float sum = 0.0f;
  for (int v = threadIdx.x; v < VV; v += 256) sum += __expf(row[v] - mx);
  red[threadIdx.x] = sum; __syncthreads();
  for (int s = 128; s > 0; s >>= 1) {
    if (threadIdx.x < s) red[threadIdx.x] += red[threadIdx.x + s];
    __syncthreads();
  }
  if (threadIdx.x == 0) {
    int tgt = y[(size_t)tnext * BB + b];
    if (tgt != 0) {
      float lp = row[tgt] - mx - __logf(red[0]);
      atomicAdd(acc, -lp);
    }
  }
}

__global__ void write_out(const float* __restrict__ acc, float* __restrict__ out)
{
  if (threadIdx.x == 0) out[0] = acc[0];
}

// ---------------------------------------------------------------------------
// Host orchestration
// ---------------------------------------------------------------------------
extern "C" void kernel_launch(void* const* d_in, const int* in_sizes, int n_in,
                              void* d_out, int out_size, void* d_ws, size_t ws_size,
                              hipStream_t stream)
{
  (void)in_sizes; (void)n_in; (void)out_size; (void)ws_size;

  const int*   y     = (const int*)  d_in[0];
  const float* ctx   = (const float*)d_in[1];
  const float* emb   = (const float*)d_in[2];
  const float* Wih0  = (const float*)d_in[3];
  const float* Whh0  = (const float*)d_in[4];
  const float* bih0  = (const float*)d_in[5];
  const float* bhh0  = (const float*)d_in[6];
  const float* Wih1  = (const float*)d_in[7];
  const float* Whh1  = (const float*)d_in[8];
  const float* bih1  = (const float*)d_in[9];
  const float* bhh1  = (const float*)d_in[10];
  const float* Wc2c  = (const float*)d_in[11];
  const float* Wh2c  = (const float*)d_in[12];
  const float* wmlp  = (const float*)d_in[13];
  const float* Wh2o  = (const float*)d_in[14];
  const float* bh2o  = (const float*)d_in[15];
  const float* Wo2p  = (const float*)d_in[16];
  const float* bo2p  = (const float*)d_in[17];

  char* ws = (char*)d_ws;
  size_t off = 0;
  auto alloc = [&](size_t bytes) -> void* {
    void* p = ws + off;
    off = (off + bytes + 255) & ~(size_t)255;
    return p;
  };

  // bf16 weights (L2-resident across the whole timestep loop)
  __bf16* wih0b = (__bf16*)alloc((size_t)3 * HH * EE * 2);
  __bf16* whh0b = (__bf16*)alloc((size_t)3 * HH * HH * 2);
  __bf16* wih1b = (__bf16*)alloc((size_t)3 * HH * CC * 2);
  __bf16* whh1b = (__bf16*)alloc((size_t)3 * HH * HH * 2);
  __bf16* wc2cb = (__bf16*)alloc((size_t)CC * CC * 2);
  __bf16* wh2cb = (__bf16*)alloc((size_t)CC * HH * 2);
  __bf16* wh2ob = (__bf16*)alloc((size_t)EE * HH * 2);
  __bf16* wo2pb = (__bf16*)alloc((size_t)VV * EE * 2);
  // precomputed tensors
  __bf16* ctxb  = (__bf16*)alloc((size_t)SS * BB * CC * 2);
  float*  ctxp  = (float*) alloc((size_t)SS * BB * CC * 4);
  __bf16* xall  = (__bf16*)alloc((size_t)(TT - 1) * BB * EE * 2);
  // per-step state / scratch
  float*  h1f  = (float*) alloc((size_t)BB * HH * 4);
  __bf16* h1b  = (__bf16*)alloc((size_t)BB * HH * 2);
  float*  hcf  = (float*) alloc((size_t)BB * HH * 4);   // current hidden (h0 -> h2 -> ...)
  __bf16* hcb  = (__bf16*)alloc((size_t)BB * HH * 2);
  float*  gi   = (float*) alloc((size_t)BB * 3 * HH * 4);
  float*  gh   = (float*) alloc((size_t)BB * 3 * HH * 4);
  float*  hid  = (float*) alloc((size_t)BB * CC * 4);
  float*  sc   = (float*) alloc((size_t)SS * BB * 4);
  __bf16* zb   = (__bf16*)alloc((size_t)BB * CC * 2);
  float*  lpre = (float*) alloc((size_t)BB * EE * 4);
  __bf16* lb   = (__bf16*)alloc((size_t)BB * EE * 2);
  float*  sbuf = (float*) alloc((size_t)BB * VV * 4);
  float*  lossacc = (float*)alloc(256);

  auto cvt = [&](const float* s, __bf16* d, int n) {
    cvt_f32_bf16<<<(n + 255) / 256, 256, 0, stream>>>(s, d, n);
  };
  // Narrow tiles: maximize wave count for small-M serial-path GEMMs.
  auto gemm1 = [&](const __bf16* A, const __bf16* W, const float* bias,
                   float* C, int M, int N, int K) {
    dim3 g(N / 16, M / 16);
    gemm_bf16_wmma<1><<<g, 32, 0, stream>>>(A, W, bias, C, M, N, K);
  };
  // Wide tiles (16x64/wave): 4x A-fragment reuse for high-parallelism GEMMs.
  auto gemm4 = [&](const __bf16* A, const __bf16* W, const float* bias,
                   float* C, int M, int N, int K) {
    dim3 g(N / 64, M / 16);
    gemm_bf16_wmma<4><<<g, 32, 0, stream>>>(A, W, bias, C, M, N, K);
  };

  // ---- one-time setup (inside graph; deterministic) ----
  cvt(Wih0, wih0b, 3 * HH * EE);
  cvt(Whh0, whh0b, 3 * HH * HH);
  cvt(Wih1, wih1b, 3 * HH * CC);
  cvt(Whh1, whh1b, 3 * HH * HH);
  cvt(Wc2c, wc2cb, CC * CC);
  cvt(Wh2c, wh2cb, CC * HH);
  cvt(Wh2o, wh2ob, EE * HH);
  cvt(Wo2p, wo2pb, VV * EE);
  cvt(ctx,  ctxb,  SS * BB * CC);

  {
    size_t n = (size_t)(TT - 1) * BB * EE;
    embed_gather<<<(unsigned)((n + 255) / 256), 256, 0, stream>>>(emb, y, xall);
  }
  init_state<<<(BB * HH + 255) / 256, 256, 0, stream>>>(hcf, hcb, lossacc);

  // ctx_p = ctx @ W_c2c^T  (8192 x 512, K=512) — off the serial path
  gemm4(ctxb, wc2cb, nullptr, ctxp, SS * BB, CC, CC);

  // ---- sequential decoder ----
  for (int t = 0; t < TT - 1; ++t) {
    // GRU cell 0
    gemm1(xall + (size_t)t * BB * EE, wih0b, bih0, gi, BB, 3 * HH, EE);
    gemm1(hcb, whh0b, bhh0, gh, BB, 3 * HH, HH);
    gru_fuse<<<(BB * HH + 255) / 256, 256, 0, stream>>>(gi, gh, hcf, h1f, h1b);

    // attention
    gemm1(h1b, wh2cb, nullptr, hid, BB, CC, HH);
    att_scores<<<SS * BB, 128, 0, stream>>>(ctxp, hid, wmlp, sc);
    att_softmax_z<<<BB, 128, 0, stream>>>(sc, ctx, zb);

    // GRU cell 1
    gemm1(zb, wih1b, bih1, gi, BB, 3 * HH, CC);
    gemm1(h1b, whh1b, bhh1, gh, BB, 3 * HH, HH);
    gru_fuse<<<(BB * HH + 255) / 256, 256, 0, stream>>>(gi, gh, h1f, hcf, hcb);

    // output head
    gemm1(hcb, wh2ob, bh2o, lpre, BB, EE, HH);
    tanh_cast<<<(BB * EE + 255) / 256, 256, 0, stream>>>(lpre, lb, BB * EE);
    gemm4(lb, wo2pb, bo2p, sbuf, BB, VV, EE);   // dominant GEMM: 64 x 32000 x 512
    loss_kernel<<<BB, 256, 0, stream>>>(sbuf, y, t + 1, lossacc);
  }

  write_out<<<1, 1, 0, stream>>>(lossacc, (float*)d_out);
}
